// Compute_Attention_22351009808472
// MI455X (gfx1250) — matrix operand order repaired and checked by reading, not compiled, not run on hardware
//
#include <hip/hip_runtime.h>

typedef __attribute__((ext_vector_type(16))) _Float16 v16h;
typedef __attribute__((ext_vector_type(8)))  _Float16 v8h;
typedef __attribute__((ext_vector_type(4)))  _Float16 v4h;
typedef __attribute__((ext_vector_type(8)))  float    v8f;
typedef __attribute__((ext_vector_type(4)))  float    v4f;

constexpr int BATCH = 4;
constexpr int HEADS = 16;
constexpr int SEQ   = 2048;
constexpr int HD    = 64;     // head dim
constexpr int QT    = 128;    // query rows per workgroup (16 per wave, 8 waves)
constexpr int NW    = 8;      // waves per workgroup
constexpr int KT    = 64;     // keys per inner iteration (amortizes softmax fixed cost)
constexpr int KP    = 72;     // LDS pitch (halves): 144B rows -> 16B aligned, 36-bank stride

static __device__ __forceinline__ v16h join16(v8h lo, v8h hi) {
  return __builtin_shufflevector(lo, hi, 0,1,2,3,4,5,6,7,8,9,10,11,12,13,14,15);
}

static __device__ __forceinline__ v8f wmma_f16(const v16h& a, const v16h& b, const v8f& c) {
  return __builtin_amdgcn_wmma_f32_16x16x32_f16(false, a, false, b, (short)0, c, false, false);
}

// xor-butterfly over the 16-lane group via ds_swizzle_b32 immediates
// (offset = xor<<10 | or<<5 | and; and=0x1f). Pattern must be an ICE.
template<int IMM>
static __device__ __forceinline__ float swz(float v) {
  return __int_as_float(__builtin_amdgcn_ds_swizzle(__float_as_int(v), IMM));
}
static __device__ __forceinline__ float grp16_max(float v) {
  v = fmaxf(v, swz<0x041F>(v));
  v = fmaxf(v, swz<0x081F>(v));
  v = fmaxf(v, swz<0x101F>(v));
  v = fmaxf(v, swz<0x201F>(v));
  return v;
}
static __device__ __forceinline__ float grp16_sum(float v) {
  v += swz<0x041F>(v);
  v += swz<0x081F>(v);
  v += swz<0x101F>(v);
  v += swz<0x201F>(v);
  return v;
}

// Process one 16(q) x 64(k) block for this wave.
template<bool MASKED>
static __device__ __forceinline__ void process_block(
    int k0, int qg, int ln, int grp,
    const _Float16* __restrict__ ldsK, const _Float16* __restrict__ ldsVt,
    _Float16* __restrict__ ldsPw,
    const v16h& aq0, const v16h& aq1,
    v8f o[4], float m[8], float l[8])
{
  // ---- scores: four 16x16 tiles (keys k0+16*kt), K-dim = HD = 64
  v8f s[4];
#pragma unroll
  for (int kt = 0; kt < 4; ++kt) {
    const _Float16* kr = &ldsK[(kt * 16 + ln) * KP + grp * 8];
    v16h b0 = join16(*(const v8h*)kr,        *(const v8h*)(kr + 16));   // d 0..31
    v16h b1 = join16(*(const v8h*)(kr + 32), *(const v8h*)(kr + 48));   // d 32..63
    v8f acc{};
    acc = wmma_f16(aq0, b0, acc);
    acc = wmma_f16(aq1, b1, acc);
    s[kt] = acc;
  }

  // ---- online softmax; row M = r + 8*grp is component r, col N = ln
  v8f av;
#pragma unroll
  for (int r = 0; r < 8; ++r) {
    if (MASKED) {
      const int q = qg + r;
#pragma unroll
      for (int kt = 0; kt < 4; ++kt)
        if (k0 + kt * 16 + ln > q) s[kt][r] = -1e30f;
    }
    const float v  = grp16_max(fmaxf(fmaxf(s[0][r], s[1][r]), fmaxf(s[2][r], s[3][r])));
    const float mn = fmaxf(m[r], v);
    av[r] = __expf(m[r] - mn);        // alpha
    m[r] = mn;
    float ps = 0.0f;
    _Float16* pp = &ldsPw[(r + 8 * grp) * KP];
#pragma unroll
    for (int kt = 0; kt < 4; ++kt) {
      const float p = __expf(s[kt][r] - mn);
      ps += p;
      pp[kt * 16 + ln] = (_Float16)p;   // bounce P through per-wave LDS (in-order)
    }
    l[r] = l[r] * av[r] + ps;           // lane-local partial
  }
#pragma unroll
  for (int t = 0; t < 4; ++t) o[t] *= av;   // vector mul -> packed f32 muls

  // ---- P A-fragments: row M = ln; frag0 keys base(e)+8g, frag1 +32
  const _Float16* pa = &ldsPw[ln * KP + grp * 8];
  v16h pf0 = join16(*(const v8h*)pa,        *(const v8h*)(pa + 16));
  v16h pf1 = join16(*(const v8h*)(pa + 32), *(const v8h*)(pa + 48));

  // ---- O += P * V  (B from V^T: N = 16t+ln; keys base(e)+8g and +32)
#pragma unroll
  for (int t = 0; t < 4; ++t) {
    const _Float16* vb = &ldsVt[(t * 16 + ln) * KP + grp * 8];
    v16h b0 = join16(*(const v8h*)vb,        *(const v8h*)(vb + 16));
    v16h b1 = join16(*(const v8h*)(vb + 32), *(const v8h*)(vb + 48));
    o[t] = wmma_f16(pf0, b0, o[t]);
    o[t] = wmma_f16(pf1, b1, o[t]);
  }
}

__global__ __launch_bounds__(256, 1)
void fa_causal_wmma(const float* __restrict__ Q, const float* __restrict__ K,
                    const float* __restrict__ V, float* __restrict__ O)
{
  __shared__ __attribute__((aligned(16))) _Float16 ldsK [KT * KP];        // [key][d]
  __shared__ __attribute__((aligned(16))) _Float16 ldsVt[HD * KP];        // [d][key]
  __shared__ __attribute__((aligned(16))) _Float16 ldsP [NW * 16 * KP];   // per-wave [row][key]

  const int tid  = threadIdx.x;
  const int wave = tid >> 5;
  const int lane = tid & 31;
  const int grp  = lane >> 4;
  const int ln   = lane & 15;

  const int bh    = blockIdx.x / (SEQ / QT);
  const int qtile = blockIdx.x % (SEQ / QT);
  const int qb    = qtile * QT;

  const float* Qp = Q + (size_t)bh * SEQ * HD;
  const float* Kp = K + (size_t)bh * SEQ * HD;
  const float* Vp = V + (size_t)bh * SEQ * HD;
  float*       Op = O + (size_t)bh * SEQ * HD;

  // ---- Q A-fragments (16-bit A 16x32 layout): M = ln, element e -> K = base(e)+8*grp
  const int qrow = qb + wave * 16 + ln;
  v16h aq0, aq1;
#pragma unroll
  for (int e = 0; e < 16; ++e) {
    const int d0 = ((e >> 3) << 4) + (e & 7) + (grp << 3);
    aq0[e] = (_Float16)(Qp[qrow * HD + d0]      * 0.125f);   // fold 1/sqrt(64)
    aq1[e] = (_Float16)(Qp[qrow * HD + d0 + 32] * 0.125f);
  }

  v8f o[4] = {v8f{}, v8f{}, v8f{}, v8f{}};
  float m[8], l[8];
#pragma unroll
  for (int r = 0; r < 8; ++r) { m[r] = -1e30f; l[r] = 0.0f; }

  // ---- software-pipelined K/V staging: prefetch tile kb+1 while computing kb
  v4f kpre[4], vpre[4];
  auto prefetch = [&](int kb) {
#pragma unroll
    for (int j = 0; j < 4; ++j) {
      const int idx = (tid + j * 256) * 4;            // flat over KT*HD f32
      kpre[j] = *(const v4f*)&Kp[(size_t)kb * KT * HD + idx];
      vpre[j] = *(const v4f*)&Vp[(size_t)kb * KT * HD + idx];
    }
  };
  auto stage = [&]() {
#pragma unroll
    for (int j = 0; j < 4; ++j) {
      const int idx = (tid + j * 256) * 4;
      const int key = idx >> 6, d = idx & 63;
      v4h kh;
#pragma unroll
      for (int e = 0; e < 4; ++e) kh[e] = (_Float16)kpre[j][e];
      *(v4h*)&ldsK[key * KP + d] = kh;                // 8B aligned
#pragma unroll
      for (int e = 0; e < 4; ++e) ldsVt[(d + e) * KP + key] = (_Float16)vpre[j][e];
    }
  };

  const int qlo = qb + wave * 16;
  const int qhi = qlo + 15;
  const int qg  = qlo + grp * 8;
  _Float16* ldsPw = &ldsP[wave * 16 * KP];

  const int nkb = qb / KT + QT / KT;     // cover keys 0 .. qb+QT-1
  prefetch(0);
  for (int kb = 0; kb < nkb; ++kb) {
    const int k0 = kb * KT;
    __syncthreads();                     // all waves done reading previous tile
    stage();
    __syncthreads();                     // tile visible to all waves
    if (kb + 1 < nkb) prefetch(kb + 1);  // overlap next tile's loads with compute
    if (k0 > qhi) continue;              // fully masked for this wave (uniform branch)
    if (k0 + KT - 1 <= qlo)
      process_block<false>(k0, qg, ln, grp, ldsK, ldsVt, ldsPw, aq0, aq1, o, m, l);
    else                                 // exactly one diagonal block per wave
      process_block<true >(k0, qg, ln, grp, ldsK, ldsVt, ldsPw, aq0, aq1, o, m, l);
  }

  // ---- epilogue: reduce lane-local sums once, normalize, coalesced store
#pragma unroll
  for (int r = 0; r < 8; ++r) {
    const float inv = 1.0f / grp16_sum(l[r]);
    const int q = qb + wave * 16 + grp * 8 + r;
    float* op = Op + (size_t)q * HD + ln;
    op[0]  = o[0][r] * inv;
    op[16] = o[1][r] * inv;
    op[32] = o[2][r] * inv;
    op[48] = o[3][r] * inv;
  }
}

extern "C" void kernel_launch(void* const* d_in, const int* in_sizes, int n_in,
                              void* d_out, int out_size, void* d_ws, size_t ws_size,
                              hipStream_t stream) {
  const float* Q = (const float*)d_in[0];
  const float* K = (const float*)d_in[1];
  const float* V = (const float*)d_in[2];
  float* O = (float*)d_out;
  const int blocks = BATCH * HEADS * (SEQ / QT);   // 1024
  fa_causal_wmma<<<blocks, 256, 0, stream>>>(Q, K, V, O);
}
